// LSTM_50663434223725
// MI455X (gfx1250) — matrix operand order repaired and checked by reading, not compiled, not run on hardware
//
#include <hip/hip_runtime.h>

#define S_LEN 512
#define I_DIM 1024
#define H_DIM 2048
#define C_DIM (H_DIM + I_DIM)   // 3072

typedef __attribute__((ext_vector_type(16))) __bf16 v16bf;
typedef __attribute__((ext_vector_type(8)))  float  v8f;

union Frag16 {              // 16 bf16 = 32 bytes = 8 VGPRs
    v16bf v;
    uint4 q[2];
};

__device__ __forceinline__ unsigned short f2bf(float f) {
    unsigned int u = __float_as_uint(f);
    u += 0x7fffu + ((u >> 16) & 1u);    // round-to-nearest-even
    return (unsigned short)(u >> 16);
}

__device__ __forceinline__ float sigmoidf_fast(float x) {
    return 1.0f / (1.0f + __expf(-x));
}

// ---------------------------------------------------------------------------
// Convert the four fp32 weight matrices (H x C) into one contiguous bf16
// buffer Wb[g][H][C] (row-major, k fastest). 96 MB fp32 -> 48 MB bf16,
// resident in the 192 MB L2 for the whole recurrence.
// ---------------------------------------------------------------------------
__global__ void convert_w_kernel(const float* __restrict__ Wf,
                                 const float* __restrict__ Wi,
                                 const float* __restrict__ Wc,
                                 const float* __restrict__ Wo,
                                 unsigned short* __restrict__ Wb) {
    const int g = blockIdx.y;
    const float* src = (g == 0) ? Wf : (g == 1) ? Wi : (g == 2) ? Wc : Wo;
    unsigned short* dst = Wb + (size_t)g * H_DIM * C_DIM;
    const size_t n = (size_t)H_DIM * C_DIM;
    for (size_t i = (size_t)blockIdx.x * blockDim.x + threadIdx.x; i < n;
         i += (size_t)gridDim.x * blockDim.x)
        dst[i] = f2bf(src[i]);
}

__global__ void convert_x_kernel(const float* __restrict__ x,
                                 unsigned short* __restrict__ xb) {
    const size_t n = (size_t)S_LEN * I_DIM;
    for (size_t i = (size_t)blockIdx.x * blockDim.x + threadIdx.x; i < n;
         i += (size_t)gridDim.x * blockDim.x)
        xb[i] = f2bf(x[i]);
}

__global__ void init_state_kernel(unsigned short* __restrict__ hb,
                                  float* __restrict__ c) {
    int i = blockIdx.x * blockDim.x + threadIdx.x;
    if (i < H_DIM) { hb[i] = 0; c[i] = 0.0f; }
}

// ---------------------------------------------------------------------------
// Phase 1: batched input projection (no serial dependency).
// Gx[g][t][m] = sum_k x[t][k] * W[g][m][H+k]   via v_wmma_f32_16x16x32_bf16.
// A = x tile (16 t x 32 k), B = W^T tile (32 k x 16 m). One wave per 16x16
// output tile, 32 K-iterations over I=1024.
// ---------------------------------------------------------------------------
__global__ void __launch_bounds__(32)
xproj_gemm_kernel(const unsigned short* __restrict__ xb,
                  const unsigned short* __restrict__ Wb,
                  float* __restrict__ Gx) {
    const int mtile = blockIdx.x;        // 0..127
    const int ttile = blockIdx.y;        // 0..31
    const int g     = blockIdx.z;        // 0..3
    const int lane  = threadIdx.x;       // 0..31
    const int half  = lane >> 4;         // lane-half selects K sub-range
    const int l15   = lane & 15;

    const int tbase = ttile * 16;
    const int mbase = mtile * 16;

    const unsigned short* xrow = xb + (size_t)(tbase + l15) * I_DIM;
    const unsigned short* wrow = Wb + (size_t)g * H_DIM * C_DIM
                               + (size_t)(mbase + l15) * C_DIM + H_DIM;

    v8f acc = {};
    for (int k = 0; k < I_DIM; k += 32) {
        Frag16 a, b;
        // A 16x32 bf16: lanes 0-15 K={0..7,16..23}, lanes 16-31 K={8..15,24..31}
        a.q[0] = *(const uint4*)(xrow + k + half * 8);
        a.q[1] = *(const uint4*)(xrow + k + half * 8 + 16);
        // B 32x16 bf16: lane = N, lanes 0-15 K={0..7,16..23}, lanes 16-31 K={8..15,24..31}
        b.q[0] = *(const uint4*)(wrow + k + half * 8);
        b.q[1] = *(const uint4*)(wrow + k + half * 8 + 16);
        acc = __builtin_amdgcn_wmma_f32_16x16x32_bf16(
                  false, a.v, false, b.v, (short)0, acc, false, false);
    }

    // D layout: lanes 0-15 N=lane, VGPR r -> M=r ; lanes 16-31 -> M=8+r
    const int m    = mbase + l15;
    const int trow = tbase + half * 8;
    float* out = Gx + ((size_t)g * S_LEN + trow) * H_DIM + m;
#pragma unroll
    for (int r = 0; r < 8; ++r)
        out[(size_t)r * H_DIM] = acc[r];
}

// ---------------------------------------------------------------------------
// Phase 2 (serial): one fused kernel per timestep.
// Block = 256 threads (8 waves) owns one 16-wide m-tile for ALL 4 gates:
//   wave w -> gate (w>>1), K-half (w&1); each wave runs 32 WMMA K-iters.
// h_{t-1} (4 KB bf16) is staged into LDS once, so the inner loop issues only
// the W stream from L2 (global) + ds_load_b128 for the B fragments.
// Tail: LDS-reduce the 8 partials, add Gx + bias, apply gates, update c,
// write h_t (bf16 double-buffered + fp32 output row).
// ---------------------------------------------------------------------------
__global__ void __launch_bounds__(256)
lstm_step_kernel(const unsigned short* __restrict__ Wb,
                 const unsigned short* __restrict__ hin,
                 unsigned short* __restrict__ hout,
                 float* __restrict__ c,
                 const float* __restrict__ Gx,
                 const float* __restrict__ bfp,
                 const float* __restrict__ bip,
                 const float* __restrict__ bcp,
                 const float* __restrict__ bop,
                 float* __restrict__ out,
                 int t) {
    __shared__ unsigned short hlds[H_DIM];   // 4 KB
    __shared__ float red[4][2][16];

    const int tid = threadIdx.x;

    // Stage h into LDS: 16 bytes per thread, fully coalesced, no divergence.
    *(uint4*)(hlds + tid * 8) = *(const uint4*)(hin + tid * 8);
    __syncthreads();

    const int mbase = blockIdx.x * 16;   // 128 blocks cover H
    const int wave  = tid >> 5;
    const int g     = wave >> 1;         // gate 0..3
    const int khalf = wave & 1;          // K-half 0..1
    const int lane  = tid & 31;
    const int half  = lane >> 4;
    const int l15   = lane & 15;

    const unsigned short* wrow = Wb + (size_t)g * H_DIM * C_DIM
                               + (size_t)(mbase + l15) * C_DIM;
    const int kstart = khalf * (H_DIM / 2);
    const int kend   = kstart + (H_DIM / 2);

    v8f acc = {};
    for (int k = kstart; k < kend; k += 32) {
        Frag16 a, b;
        a.q[0] = *(const uint4*)(wrow + k + half * 8);        // W: global (L2)
        a.q[1] = *(const uint4*)(wrow + k + half * 8 + 16);
        b.q[0] = *(const uint4*)(hlds + k + half * 8);        // h: LDS
        b.q[1] = *(const uint4*)(hlds + k + half * 8 + 16);
        acc = __builtin_amdgcn_wmma_f32_16x16x32_bf16(
                  false, a.v, false, b.v, (short)0, acc, false, false);
    }

    // All 16 columns of D are identical (broadcast B).
    // Lane 0 holds y[mbase+0..7], lane 16 holds y[mbase+8..15].
    if (lane == 0) {
#pragma unroll
        for (int r = 0; r < 8; ++r) red[g][khalf][r] = acc[r];
    }
    if (lane == 16) {
#pragma unroll
        for (int r = 0; r < 8; ++r) red[g][khalf][8 + r] = acc[r];
    }
    __syncthreads();

    if (tid < 16) {
        const int m = mbase + tid;
        float pf = red[0][0][tid] + red[0][1][tid]
                 + Gx[((size_t)0 * S_LEN + t) * H_DIM + m] + bfp[m];
        float pi = red[1][0][tid] + red[1][1][tid]
                 + Gx[((size_t)1 * S_LEN + t) * H_DIM + m] + bip[m];
        float pc = red[2][0][tid] + red[2][1][tid]
                 + Gx[((size_t)2 * S_LEN + t) * H_DIM + m] + bcp[m];
        float po = red[3][0][tid] + red[3][1][tid]
                 + Gx[((size_t)3 * S_LEN + t) * H_DIM + m] + bop[m];

        float fg = sigmoidf_fast(pf);
        float ig = sigmoidf_fast(pi);
        float ct = tanhf(pc);
        float og = sigmoidf_fast(po);

        float cn = fg * c[m] + ig * ct;
        float hn = og * tanhf(cn);

        c[m]    = cn;
        hout[m] = f2bf(hn);
        out[(size_t)t * H_DIM + m] = hn;
        if (t == S_LEN - 1) {
            out[(size_t)S_LEN * H_DIM + m]         = hn;   // h_T
            out[(size_t)S_LEN * H_DIM + H_DIM + m] = cn;   // c_T
        }
    }
}

// ---------------------------------------------------------------------------
extern "C" void kernel_launch(void* const* d_in, const int* in_sizes, int n_in,
                              void* d_out, int out_size, void* d_ws, size_t ws_size,
                              hipStream_t stream) {
    const float* x  = (const float*)d_in[0];
    const float* Wf = (const float*)d_in[1];
    const float* bf = (const float*)d_in[2];
    const float* Wi = (const float*)d_in[3];
    const float* bi = (const float*)d_in[4];
    const float* Wc = (const float*)d_in[5];
    const float* bc = (const float*)d_in[6];
    const float* Wo = (const float*)d_in[7];
    const float* bo = (const float*)d_in[8];
    float* out = (float*)d_out;

    char* ws = (char*)d_ws;
    size_t off = 0;
    auto alloc = [&](size_t bytes) -> void* {
        void* p = ws + off;
        off = (off + bytes + 255) & ~(size_t)255;
        return p;
    };
    unsigned short* Wb  = (unsigned short*)alloc((size_t)4 * H_DIM * C_DIM * 2); // 48 MB bf16 weights
    unsigned short* xb  = (unsigned short*)alloc((size_t)S_LEN * I_DIM * 2);     // 1 MB bf16 x
    float*          Gx  = (float*)alloc((size_t)4 * S_LEN * H_DIM * 4);          // 16 MB x-projection
    unsigned short* hA  = (unsigned short*)alloc((size_t)H_DIM * 2);             // h ping
    unsigned short* hB  = (unsigned short*)alloc((size_t)H_DIM * 2);             // h pong
    float*          cst = (float*)alloc((size_t)H_DIM * 4);                      // c state

    // One-time prep
    convert_w_kernel<<<dim3(512, 4), dim3(256), 0, stream>>>(Wf, Wi, Wc, Wo, Wb);
    convert_x_kernel<<<dim3(512), dim3(256), 0, stream>>>(x, xb);
    init_state_kernel<<<dim3(8), dim3(256), 0, stream>>>(hA, cst);

    // Phase 1: batched input projection for all 512 timesteps (WMMA GEMM)
    xproj_gemm_kernel<<<dim3(128, 32, 4), dim3(32), 0, stream>>>(xb, Wb, Gx);

    // Phase 2: serial recurrence, one fused kernel per step (stream-ordered),
    // h double-buffered across steps.
    for (int t = 0; t < S_LEN; ++t) {
        const unsigned short* hin  = (t & 1) ? hB : hA;
        unsigned short*       hout = (t & 1) ? hA : hB;
        lstm_step_kernel<<<dim3(128), dim3(256), 0, stream>>>(
            Wb, hin, hout, cst, Gx, bf, bi, bc, bo, out, t);
    }
}
